// CGCNN_3908420239767
// MI455X (gfx1250) — matrix operand order repaired and checked by reading, not compiled, not run on hardware
//
#include <hip/hip_runtime.h>
#include <cstdint>
#include <cstddef>

#define Hdim 128
#define Nn   50000
#define Ee   500000
#define Bb   64
#define NDd  12
#define CDd  71
#define EPSf 1e-5f

typedef unsigned short ush;
typedef __attribute__((ext_vector_type(16))) __bf16 v16bf;
typedef __attribute__((ext_vector_type(8)))  float  v8f;
typedef __attribute__((ext_vector_type(4)))  unsigned int v4u;

union FragBF { v16bf v; v4u q[2]; ush u[16]; };

__device__ inline float bf2f(ush u) {
    union { uint32_t i; float f; } c; c.i = ((uint32_t)u) << 16; return c.f;
}
__device__ inline ush f2bf(float f) {
    union { float f; uint32_t i; } c; c.f = f;
    uint32_t x = c.i + 0x7FFFu + ((c.i >> 16) & 1u);
    return (ush)(x >> 16);
}
__device__ inline float softplusf(float x) {
    return x > 20.f ? x : log1pf(__expf(x));
}

// ---------------------------------------------------------------------------
// D-tile (16 rows x 128 cols per wave, fragment layout) -> LDS -> row-major
// bf16 global store. bias added here (z = A@W + b, pre-BN).
// ---------------------------------------------------------------------------
__device__ inline void store_strip(v8f* acc, const float* __restrict__ bias,
                                   ush* ldsWave, ush* __restrict__ out,
                                   int rowBase, int R, int lane)
{
    int nBase = lane & 15;
    int mBase = (lane & 16) >> 1;   // 0 or 8
#pragma unroll
    for (int nt = 0; nt < 8; ++nt) {
#pragma unroll
        for (int r = 0; r < 8; ++r) {
            int n = nt * 16 + nBase;
            ldsWave[(mBase + r) * Hdim + n] = f2bf(acc[nt][r] + bias[n]);
        }
    }
    __syncthreads();
    int lr = lane >> 1;             // 0..15  (row within strip)
    int lc = (lane & 1) * 64;       // 0 or 64
    int gRow = rowBase + lr;
    if (gRow < R) {
        const ush* s = ldsWave + lr * Hdim + lc;
        ush* d = out + (size_t)gRow * Hdim + lc;
#pragma unroll
        for (int i = 0; i < 8; ++i)
            reinterpret_cast<v4u*>(d)[i] = reinterpret_cast<const v4u*>(s)[i];
    }
}

// ---------------------------------------------------------------------------
// Fused gather GEMM: z[e,:] = concat(h[row[e]], e[e], h[col[e]]) @ W + b
// K = 384 (12 chunks of 32), N = 128 (8 WMMA n-tiles), 16 edges per wave.
// All 8 B-fragments are preloaded per K-chunk so the compiler emits one load
// clause + single s_wait, then 8 back-to-back v_wmma.
// ---------------------------------------------------------------------------
__global__ __launch_bounds__(256) void kgemm_ef(
    const ush* __restrict__ hbf, const ush* __restrict__ ebf,
    const int* __restrict__ eidx,
    const ush* __restrict__ wsw, const float* __restrict__ bias,
    ush* __restrict__ out)
{
    __shared__ ush lds[8][16 * Hdim];
    int wave = threadIdx.x >> 5, lane = threadIdx.x & 31;
    int rowBase = blockIdx.x * 128 + wave * 16;
    int myRow = rowBase + (lane & 15);
    int e = myRow < Ee ? myRow : 0;
    int rsrc = eidx[e];
    int cdst = eidx[Ee + e];
    int off = (lane & 16) >> 1;     // 0 or 8

    v8f acc[8];
#pragma unroll
    for (int nt = 0; nt < 8; ++nt)
#pragma unroll
        for (int r = 0; r < 8; ++r) acc[nt][r] = 0.f;

#pragma unroll
    for (int sgi = 0; sgi < 3; ++sgi) {
        const ush* base = (sgi == 0) ? hbf + (size_t)rsrc * Hdim
                        : (sgi == 1) ? ebf + (size_t)e    * Hdim
                                     : hbf + (size_t)cdst * Hdim;
        for (int kk = 0; kk < 4; ++kk) {
            int kc = sgi * 4 + kk;
            FragBF a;
            {
                const ush* s = base + (kk * 32 + off);
                a.q[0] = *reinterpret_cast<const v4u*>(s);
                a.q[1] = *reinterpret_cast<const v4u*>(s + 16);
            }
            const ush* wb = wsw + (size_t)(kc * 8) * 512 + lane * 16;
            FragBF b[8];
#pragma unroll
            for (int nt = 0; nt < 8; ++nt) {
                b[nt].q[0] = *reinterpret_cast<const v4u*>(wb + nt * 512);
                b[nt].q[1] = *reinterpret_cast<const v4u*>(wb + nt * 512 + 8);
            }
#pragma unroll
            for (int nt = 0; nt < 8; ++nt)
                acc[nt] = __builtin_amdgcn_wmma_f32_16x16x32_bf16(
                    false, a.v, false, b[nt].v, (short)0, acc[nt], false, false);
        }
    }
    store_strip(acc, bias, lds[wave], out, rowBase, Ee, lane);
}

// ---------------------------------------------------------------------------
// Dense GEMM: z = A(bf16 row-major [R,128]) @ W + b   (K = 128, 4 chunks)
// ---------------------------------------------------------------------------
__global__ __launch_bounds__(256) void kgemm_dense(
    const ush* __restrict__ A, int R,
    const ush* __restrict__ wsw, const float* __restrict__ bias,
    ush* __restrict__ out)
{
    __shared__ ush lds[8][16 * Hdim];
    int wave = threadIdx.x >> 5, lane = threadIdx.x & 31;
    int rowBase = blockIdx.x * 128 + wave * 16;
    int myRow = rowBase + (lane & 15);
    int ar = myRow < R ? myRow : R - 1;
    const ush* base = A + (size_t)ar * Hdim;
    int off = (lane & 16) >> 1;

    v8f acc[8];
#pragma unroll
    for (int nt = 0; nt < 8; ++nt)
#pragma unroll
        for (int r = 0; r < 8; ++r) acc[nt][r] = 0.f;

#pragma unroll
    for (int kc = 0; kc < 4; ++kc) {
        FragBF a;
        {
            const ush* s = base + (kc * 32 + off);
            a.q[0] = *reinterpret_cast<const v4u*>(s);
            a.q[1] = *reinterpret_cast<const v4u*>(s + 16);
        }
        const ush* wb = wsw + (size_t)(kc * 8) * 512 + lane * 16;
        FragBF b[8];
#pragma unroll
        for (int nt = 0; nt < 8; ++nt) {
            b[nt].q[0] = *reinterpret_cast<const v4u*>(wb + nt * 512);
            b[nt].q[1] = *reinterpret_cast<const v4u*>(wb + nt * 512 + 8);
        }
#pragma unroll
        for (int nt = 0; nt < 8; ++nt)
            acc[nt] = __builtin_amdgcn_wmma_f32_16x16x32_bf16(
                false, a.v, false, b[nt].v, (short)0, acc[nt], false, false);
    }
    store_strip(acc, bias, lds[wave], out, rowBase, R, lane);
}

// ---------------------------------------------------------------------------
// Weight pre-swizzle: row-major f32 [K,128] -> bf16 fragment order
// dst[((kc*8+nt)*32+lane)*16 + j] = W[kc*32 + (lane&16)/2 + j + (j>=8 ? 8 : 0)][nt*16 + (lane&15)]
// ---------------------------------------------------------------------------
__global__ void kswz(const float* __restrict__ W, ush* __restrict__ dst, int K)
{
    int t = blockIdx.x * 256 + threadIdx.x;
    if (t >= K * 8) return;
    int lane = t & 31;
    int nt = (t >> 5) & 7;
    int kc = t >> 8;
    ush* d = dst + (size_t)t * 16;
    int kb = kc * 32 + ((lane & 16) >> 1);
    int cb = nt * 16 + (lane & 15);
    for (int j = 0; j < 16; ++j)
        d[j] = f2bf(W[(size_t)(kb + j + (j >= 8 ? 8 : 0)) * 128 + cb]);
}

// ---------------------------------------------------------------------------
// BatchNorm column statistics over bf16 [R,128]: sums[0:128]=sum, [128:256]=sumsq
// ---------------------------------------------------------------------------
__global__ __launch_bounds__(256) void kstats(const ush* __restrict__ z, int R,
                                              float* __restrict__ sums)
{
    __shared__ float ls[512];
    int t = threadIdx.x;
    int c = t & 127;
    int half = t >> 7;
    float s = 0.f, s2 = 0.f;
    int base = blockIdx.x * 512;
    for (int r = half; r < 512; r += 2) {
        int row = base + r;
        if (row < R) {
            float v = bf2f(z[(size_t)row * 128 + c]);
            s += v; s2 += v * v;
        }
    }
    ls[t] = s; ls[256 + t] = s2;
    __syncthreads();
    if (t < 128) {
        unsafeAtomicAdd(&sums[t],       ls[t] + ls[t + 128]);
        unsafeAtomicAdd(&sums[128 + t], ls[256 + t] + ls[256 + t + 128]);
    }
}

__global__ void kfin(const float* __restrict__ sums, float Rinv,
                     const float* __restrict__ g, const float* __restrict__ beta,
                     float* __restrict__ ss)
{
    int c = threadIdx.x;  // 128 threads
    float m = sums[c] * Rinv;
    float v = sums[128 + c] * Rinv - m * m;
    float sc = g[c] * rsqrtf(v + EPSf);
    ss[c] = sc;
    ss[128 + c] = beta[c] - m * sc;
}

// y = softplus(z*scale + shift); writes bf16 (and optionally f32 copy)
__global__ __launch_bounds__(256) void kapply(const ush* __restrict__ z,
                                              const float* __restrict__ ss, int total,
                                              ush* __restrict__ obf, float* __restrict__ of32)
{
    int i = blockIdx.x * 256 + threadIdx.x;
    if (i >= total) return;
    int c = i & 127;
    float y = softplusf(bf2f(z[i]) * ss[c] + ss[128 + c]);
    obf[i] = f2bf(y);
    if (of32) of32[i] = y;
}

__global__ void kzero(float* __restrict__ p, int n)
{
    int i = blockIdx.x * 256 + threadIdx.x;
    if (i < n) p[i] = 0.f;
}

// node embed z = x@node_W + b   (K=12, scalar)
__global__ __launch_bounds__(256) void knode(const float* __restrict__ x,
                                             const float* __restrict__ W,
                                             const float* __restrict__ b,
                                             ush* __restrict__ out)
{
    int i = blockIdx.x * 256 + threadIdx.x;
    if (i >= Nn * Hdim) return;
    int n = i >> 7, k = i & 127;
    float a = b[k];
    for (int d = 0; d < NDd; ++d) a += x[n * NDd + d] * W[d * 128 + k];
    out[i] = f2bf(a);
}

// edge embed z = edge_attr@edge_W + b   (K=1)
__global__ __launch_bounds__(256) void kedge(const float* __restrict__ ea,
                                             const float* __restrict__ W,
                                             const float* __restrict__ b,
                                             ush* __restrict__ out)
{
    int i = blockIdx.x * 256 + threadIdx.x;
    if (i >= Ee * Hdim) return;
    int e = i >> 7, k = i & 127;
    out[i] = f2bf(ea[e] * W[k] + b[k]);
}

__global__ __launch_bounds__(256) void kscatter(const ush* __restrict__ msg,
                                                const int* __restrict__ col,
                                                float* __restrict__ agg)
{
    int i = blockIdx.x * 256 + threadIdx.x;
    if (i >= Ee * Hdim) return;
    int e = i >> 7, k = i & 127;
    unsafeAtomicAdd(&agg[(size_t)col[e] * 128 + k], bf2f(msg[i]));
}

__global__ __launch_bounds__(256) void kresid(const float* __restrict__ hf,
                                              const float* __restrict__ agg,
                                              ush* __restrict__ tbf)
{
    int i = blockIdx.x * 256 + threadIdx.x;
    if (i >= Nn * Hdim) return;
    tbf[i] = f2bf(hf[i] + agg[i]);
}

__global__ __launch_bounds__(256) void kpool(const float* __restrict__ hf,
                                             const int* __restrict__ batch,
                                             float* __restrict__ pool,
                                             float* __restrict__ cnt)
{
    int i = blockIdx.x * 256 + threadIdx.x;
    if (i >= Nn * Hdim) return;
    int n = i >> 7, k = i & 127;
    int bb = batch[n];
    unsafeAtomicAdd(&pool[bb * 128 + k], hf[i]);
    if (k == 0) unsafeAtomicAdd(&cnt[bb], 1.f);
}

// ---------------------------------------------------------------------------
// Tail: gm finalize + comp block + fc0 + fc1 + out   (single workgroup, B=64)
// ---------------------------------------------------------------------------
__global__ __launch_bounds__(256) void kfinal(
    const float* __restrict__ pool, const float* __restrict__ cnt,
    const float* __restrict__ compF,
    const float* __restrict__ compW, const float* __restrict__ compB,
    const float* __restrict__ compG, const float* __restrict__ compBeta,
    const float* __restrict__ fc0W, const float* __restrict__ fc0B,
    const float* __restrict__ fc0G, const float* __restrict__ fc0Beta,
    const float* __restrict__ fc1W, const float* __restrict__ fc1B,
    const float* __restrict__ fc1G, const float* __restrict__ fc1Beta,
    const float* __restrict__ outW, const float* __restrict__ outB,
    float* __restrict__ out)
{
    __shared__ float X[Bb * Hdim];
    __shared__ float Y[Bb * Hdim];
    int t = threadIdx.x;

    // z1 = comp_features @ comp_W + b   -> X
    for (int i = t; i < Bb * Hdim; i += 256) {
        int r = i >> 7, c = i & 127;
        float a = compB[c];
        for (int k = 0; k < CDd; ++k) a += compF[r * CDd + k] * compW[k * 128 + c];
        X[i] = a;
    }
    __syncthreads();
    if (t < 128) {          // stats over X -> scale/shift in Y[0:256]
        float s = 0.f, s2 = 0.f;
        for (int r = 0; r < Bb; ++r) { float v = X[r * 128 + t]; s += v; s2 += v * v; }
        float m = s * (1.f / Bb), var = s2 * (1.f / Bb) - m * m;
        float sc = compG[t] * rsqrtf(var + EPSf);
        Y[t] = sc; Y[128 + t] = compBeta[t] - m * sc;
    }
    __syncthreads();
    for (int i = t; i < Bb * Hdim; i += 256) {   // X = c = softplus(bn(z1))
        int c = i & 127;
        X[i] = softplusf(X[i] * Y[c] + Y[128 + c]);
    }
    __syncthreads();
    // z2 = [gm | c] @ fc0_W + b  -> Y
    for (int i = t; i < Bb * Hdim; i += 256) {
        int r = i >> 7, c = i & 127;
        float a = fc0B[c];
        float icnt = 1.f / fmaxf(cnt[r], 1.f);
        for (int k = 0; k < 128; ++k) a += (pool[r * 128 + k] * icnt) * fc0W[k * 128 + c];
        for (int k = 0; k < 128; ++k) a += X[r * 128 + k] * fc0W[(128 + k) * 128 + c];
        Y[i] = a;
    }
    __syncthreads();
    if (t < 128) {          // stats over Y -> scale/shift in X[0:256]
        float s = 0.f, s2 = 0.f;
        for (int r = 0; r < Bb; ++r) { float v = Y[r * 128 + t]; s += v; s2 += v * v; }
        float m = s * (1.f / Bb), var = s2 * (1.f / Bb) - m * m;
        float sc = fc0G[t] * rsqrtf(var + EPSf);
        X[t] = sc; X[128 + t] = fc0Beta[t] - m * sc;
    }
    __syncthreads();
    for (int i = t; i < Bb * Hdim; i += 256) {
        int c = i & 127;
        Y[i] = softplusf(Y[i] * X[c] + X[128 + c]);
    }
    __syncthreads();
    // z3 = g2 @ fc1_W + b -> X
    for (int i = t; i < Bb * Hdim; i += 256) {
        int r = i >> 7, c = i & 127;
        float a = fc1B[c];
        for (int k = 0; k < 128; ++k) a += Y[r * 128 + k] * fc1W[k * 128 + c];
        X[i] = a;
    }
    __syncthreads();
    if (t < 128) {          // stats over X -> scale/shift in Y[0:256]
        float s = 0.f, s2 = 0.f;
        for (int r = 0; r < Bb; ++r) { float v = X[r * 128 + t]; s += v; s2 += v * v; }
        float m = s * (1.f / Bb), var = s2 * (1.f / Bb) - m * m;
        float sc = fc1G[t] * rsqrtf(var + EPSf);
        Y[t] = sc; Y[128 + t] = fc1Beta[t] - m * sc;
    }
    __syncthreads();
    for (int i = t; i < Bb * Hdim; i += 256) {
        int c = i & 127;
        X[i] = softplusf(X[i] * Y[c] + Y[128 + c]);
    }
    __syncthreads();
    if (t < Bb) {
        float a = outB[0];
        for (int k = 0; k < 128; ++k) a += X[t * 128 + k] * outW[k];
        out[t] = a;
    }
}

// ---------------------------------------------------------------------------
extern "C" void kernel_launch(void* const* d_in, const int* in_sizes, int n_in,
                              void* d_out, int out_size, void* d_ws, size_t ws_size,
                              hipStream_t stream)
{
    (void)in_sizes; (void)n_in; (void)out_size; (void)ws_size;

    const float* x       = (const float*)d_in[0];
    const int*   eidx    = (const int*)  d_in[1];
    const float* eattr   = (const float*)d_in[2];
    const int*   batch   = (const int*)  d_in[3];
    const float* compF   = (const float*)d_in[4];
    const float* node_W  = (const float*)d_in[5];
    const float* node_b  = (const float*)d_in[6];
    const float* node_g  = (const float*)d_in[7];
    const float* node_be = (const float*)d_in[8];
    const float* edge_W  = (const float*)d_in[9];
    const float* edge_b  = (const float*)d_in[10];
    const float* edge_g  = (const float*)d_in[11];
    const float* edge_be = (const float*)d_in[12];
    const float* ce1_W   = (const float*)d_in[13];
    const float* ce1_b   = (const float*)d_in[14];
    const float* ce1_g   = (const float*)d_in[15];
    const float* ce1_be  = (const float*)d_in[16];
    const float* ce2_W   = (const float*)d_in[17];
    const float* ce2_b   = (const float*)d_in[18];
    const float* ce2_g   = (const float*)d_in[19];
    const float* ce2_be  = (const float*)d_in[20];
    const float* cn_W    = (const float*)d_in[21];
    const float* cn_b    = (const float*)d_in[22];
    const float* cn_g    = (const float*)d_in[23];
    const float* cn_be   = (const float*)d_in[24];
    const float* comp_W  = (const float*)d_in[25];
    const float* comp_b  = (const float*)d_in[26];
    const float* comp_g  = (const float*)d_in[27];
    const float* comp_be = (const float*)d_in[28];
    const float* fc0_W   = (const float*)d_in[29];
    const float* fc0_b   = (const float*)d_in[30];
    const float* fc0_g   = (const float*)d_in[31];
    const float* fc0_be  = (const float*)d_in[32];
    const float* fc1_W   = (const float*)d_in[33];
    const float* fc1_b   = (const float*)d_in[34];
    const float* fc1_g   = (const float*)d_in[35];
    const float* fc1_be  = (const float*)d_in[36];
    const float* out_W   = (const float*)d_in[37];
    const float* out_b   = (const float*)d_in[38];
    float* outp = (float*)d_out;

    // ---- workspace carve ----
    char* W = (char*)d_ws;
    size_t cur = 0;
    auto take = [&](size_t bytes) -> void* {
        void* p = W + cur;
        cur = (cur + bytes + 255) & ~(size_t)255;
        return p;
    };
    float* hf32 = (float*)take((size_t)Nn * Hdim * 4);
    ush*   hbf  = (ush*)  take((size_t)Nn * Hdim * 2);
    ush*   tbf  = (ush*)  take((size_t)Nn * Hdim * 2);
    ush*   ebf  = (ush*)  take((size_t)Ee * Hdim * 2);
    ush*   bufA = (ush*)  take((size_t)Ee * Hdim * 2);
    ush*   bufB = (ush*)  take((size_t)Ee * Hdim * 2);
    float* agg  = (float*)take((size_t)Nn * Hdim * 4);
    ush*   ce1w = (ush*)  take((size_t)4 * 49152 * 2);  // 4 x (384/32)*8*32*16
    ush*   ce2w = (ush*)  take((size_t)4 * 16384 * 2);
    ush*   cnw  = (ush*)  take((size_t)4 * 16384 * 2);
    float* sums = (float*)take(256 * 4);
    float* ss   = (float*)take(256 * 4);
    float* pool = (float*)take((size_t)Bb * Hdim * 4);
    float* cnt  = (float*)take((size_t)Bb * 4);

    const int GE = (Ee * Hdim) / 256;          // 250000
    const int GN = (Nn * Hdim) / 256;          // 25000
    const int gemmE = (Ee + 127) / 128;        // 3907
    const int gemmN = (Nn + 127) / 128;        // 391
    const int statsE = (Ee + 511) / 512;       // 977
    const int statsN = (Nn + 511) / 512;       // 98

    // ---- weight pre-swizzle (bf16 fragment order) ----
    for (int i = 0; i < 4; ++i) {
        kswz<<<(384 * 8 + 255) / 256, 256, 0, stream>>>(ce1_W + (size_t)i * 384 * 128, ce1w + (size_t)i * 49152, 384);
        kswz<<<(128 * 8 + 255) / 256, 256, 0, stream>>>(ce2_W + (size_t)i * 128 * 128, ce2w + (size_t)i * 16384, 128);
        kswz<<<(128 * 8 + 255) / 256, 256, 0, stream>>>(cn_W  + (size_t)i * 128 * 128, cnw  + (size_t)i * 16384, 128);
    }

    // ---- node embedding: h = softplus(BN(x@W+b)) ----
    knode<<<GN, 256, 0, stream>>>(x, node_W, node_b, bufA);
    kzero<<<1, 256, 0, stream>>>(sums, 256);
    kstats<<<statsN, 256, 0, stream>>>(bufA, Nn, sums);
    kfin<<<1, 128, 0, stream>>>(sums, 1.f / Nn, node_g, node_be, ss);
    kapply<<<GN, 256, 0, stream>>>(bufA, ss, Nn * Hdim, hbf, hf32);

    // ---- edge embedding: e = softplus(BN(ea@W+b)) ----
    kedge<<<GE, 256, 0, stream>>>(eattr, edge_W, edge_b, bufA);
    kzero<<<1, 256, 0, stream>>>(sums, 256);
    kstats<<<statsE, 256, 0, stream>>>(bufA, Ee, sums);
    kfin<<<1, 128, 0, stream>>>(sums, 1.f / Ee, edge_g, edge_be, ss);
    kapply<<<GE, 256, 0, stream>>>(bufA, ss, Ee * Hdim, ebf, nullptr);

    // ---- 4 x CGConv ----
    for (int i = 0; i < 4; ++i) {
        // msg = block(concat(h[row], e, h[col]) @ ce1)
        kgemm_ef<<<gemmE, 256, 0, stream>>>(hbf, ebf, eidx, ce1w + (size_t)i * 49152, ce1_b + i * 128, bufA);
        kzero<<<1, 256, 0, stream>>>(sums, 256);
        kstats<<<statsE, 256, 0, stream>>>(bufA, Ee, sums);
        kfin<<<1, 128, 0, stream>>>(sums, 1.f / Ee, ce1_g + i * 128, ce1_be + i * 128, ss);
        kapply<<<GE, 256, 0, stream>>>(bufA, ss, Ee * Hdim, bufB, nullptr);

        // msg = block(msg @ ce2)
        kgemm_dense<<<gemmE, 256, 0, stream>>>(bufB, Ee, ce2w + (size_t)i * 16384, ce2_b + i * 128, bufA);
        kzero<<<1, 256, 0, stream>>>(sums, 256);
        kstats<<<statsE, 256, 0, stream>>>(bufA, Ee, sums);
        kfin<<<1, 128, 0, stream>>>(sums, 1.f / Ee, ce2_g + i * 128, ce2_be + i * 128, ss);
        kapply<<<GE, 256, 0, stream>>>(bufA, ss, Ee * Hdim, bufB, nullptr);

        // agg = segment_sum(msg, col)
        kzero<<<(Nn * Hdim + 255) / 256, 256, 0, stream>>>(agg, Nn * Hdim);
        kscatter<<<GE, 256, 0, stream>>>(bufB, eidx + Ee, agg);

        // h = block((h + agg) @ cn)
        kresid<<<GN, 256, 0, stream>>>(hf32, agg, tbf);
        kgemm_dense<<<gemmN, 256, 0, stream>>>(tbf, Nn, cnw + (size_t)i * 16384, cn_b + i * 128, bufA);
        kzero<<<1, 256, 0, stream>>>(sums, 256);
        kstats<<<statsN, 256, 0, stream>>>(bufA, Nn, sums);
        kfin<<<1, 128, 0, stream>>>(sums, 1.f / Nn, cn_g + i * 128, cn_be + i * 128, ss);
        kapply<<<GN, 256, 0, stream>>>(bufA, ss, Nn * Hdim, hbf, hf32);
    }

    // ---- mean pool + tail MLP ----
    kzero<<<(Bb * Hdim + 255) / 256, 256, 0, stream>>>(pool, Bb * Hdim);
    kzero<<<1, 256, 0, stream>>>(cnt, Bb);
    kpool<<<GN, 256, 0, stream>>>(hf32, batch, pool, cnt);

    kfinal<<<1, 256, 0, stream>>>(pool, cnt, compF,
                                  comp_W, comp_b, comp_g, comp_be,
                                  fc0_W, fc0_b, fc0_g, fc0_be,
                                  fc1_W, fc1_b, fc1_g, fc1_be,
                                  out_W, out_b, outp);
}
